// SpikeNeuralNetwork_74010876445233
// MI455X (gfx1250) — matrix operand order repaired and checked by reading, not compiled, not run on hardware
//
#include <hip/hip_runtime.h>

typedef __attribute__((ext_vector_type(16))) _Float16 v16h;
typedef __attribute__((ext_vector_type(8)))  _Float16 v8h;
typedef __attribute__((ext_vector_type(8)))  float    v8f;
typedef __attribute__((ext_vector_type(4)))  int      v4i_;

#define B_    4096
#define DIN_  512
#define DH_   2048
#define DOUT_ 512
#define T_    32
#define BETA_ 0.95f
#define THR_  1.0f
#define LDW_  2056   // DH_ + 8 halves pad: row stride 4112B breaks LDS bank alignment

#ifndef __has_builtin
#define __has_builtin(x) 0
#endif

#if __has_builtin(__builtin_amdgcn_global_load_async_to_lds_b128) && \
    __has_builtin(__builtin_amdgcn_s_wait_asynccnt)
#define HAVE_ASYNC_LDS 1
typedef __attribute__((address_space(1))) v4i_ as1_v4i;
typedef __attribute__((address_space(3))) v4i_ as3_v4i;
#else
#define HAVE_ASYNC_LDS 0
#endif

union ABfrag { v16h v; v8h h[2]; };

__device__ inline v8f vzero8() {
  v8f z;
#pragma unroll
  for (int e = 0; e < 8; ++e) z[e] = 0.f;
  return z;
}

// ---------------- f32 -> f16 conversion (streaming, read-once -> NT loads) ----
__global__ void cvt_f16(const float* __restrict__ in, _Float16* __restrict__ out, int n) {
  int i = blockIdx.x * blockDim.x + threadIdx.x;
  if (i < n) out[i] = (_Float16)__builtin_nontemporal_load(in + i);
}

// ---------------- fc1: cur1 = x @ W1^T + b1 (f16 out) ----------------
__global__ __launch_bounds__(256) void fc1_kernel(const _Float16* __restrict__ xh,
                                                  const _Float16* __restrict__ w1h,
                                                  const float* __restrict__ b1,
                                                  _Float16* __restrict__ cur1h) {
  const int blk  = blockIdx.x;          // 16-row M tile
  const int wave = threadIdx.x >> 5;
  const int lane = threadIdx.x & 31;
  const int lm   = lane & 15;
  const int hi   = lane >> 4;
  const int m0   = blk * 16;

  const _Float16* arow = xh + (size_t)(m0 + lm) * DIN_;

#pragma unroll 1
  for (int i = 0; i < 16; ++i) {        // 8 waves x 16 tiles = 128 N-tiles = 2048 cols
    const int n = (wave * 16 + i) * 16 + lm;
    const _Float16* brow = w1h + (size_t)n * DIN_;
    v8f acc = vzero8();
#pragma unroll
    for (int k0 = 0; k0 < DIN_; k0 += 32) {
      ABfrag a, b;
      a.h[0] = *(const v8h*)(arow + k0 + hi * 8);
      a.h[1] = *(const v8h*)(arow + k0 + 16 + hi * 8);
      b.h[0] = *(const v8h*)(brow + k0 + hi * 8);
      b.h[1] = *(const v8h*)(brow + k0 + 16 + hi * 8);
      acc = __builtin_amdgcn_wmma_f32_16x16x32_f16(false, a.v, false, b.v,
                                                   (short)0, acc, false, false);
    }
    const float bn = b1[n];
#pragma unroll
    for (int r = 0; r < 8; ++r) {
      const int gm = m0 + r + 8 * hi;   // D-frag row mapping
      cur1h[(size_t)gm * DH_ + n] = (_Float16)(acc[r] + bn);
    }
  }
}

// ---------------- fused 32-step SNN recurrence ----------------
__global__ __launch_bounds__(256) void snn_kernel(const _Float16* __restrict__ cur1h,
                                                  const _Float16* __restrict__ w2h,
                                                  const float* __restrict__ b2,
                                                  float* __restrict__ out_spk,
                                                  float* __restrict__ out_mem) {
  __shared__ __align__(16) _Float16 spk[16 * LDW_];   // spike tile, WMMA-A layout source
  __shared__ __align__(16) _Float16 c1s[16 * LDW_];   // constant fc1 currents (f16)

  const int tid = threadIdx.x;
  const int blk = blockIdx.x;
  const int m0  = blk * 16;

  // elementwise ownership: thread -> (row r, 128-col chunk)
  const int er = tid & 15;
  const int ec = (tid >> 4) * 128;

  // stage cur1 tile into LDS once (constant across time)
  {
    const _Float16* src = cur1h + (size_t)(m0 + er) * DH_ + ec;
    _Float16* dst = c1s + er * LDW_ + ec;
#if HAVE_ASYNC_LDS
#pragma unroll
    for (int j = 0; j < 16; ++j)
      __builtin_amdgcn_global_load_async_to_lds_b128(
          (as1_v4i*)(src + j * 8), (as3_v4i*)(dst + j * 8), 0, 0);
    __builtin_amdgcn_s_wait_asynccnt(0);
#else
#pragma unroll
    for (int j = 0; j < 16; ++j)
      *(v8h*)(dst + j * 8) = *(const v8h*)(src + j * 8);
#endif
  }

  float mem1v[128];
#pragma unroll
  for (int j = 0; j < 128; ++j) mem1v[j] = 0.f;

  const int wave = tid >> 5;
  const int lane = tid & 31;
  const int lm   = lane & 15;
  const int hi   = lane >> 4;

  float mem2v[4][8];
  float b2v[4];
#pragma unroll
  for (int i = 0; i < 4; ++i) {
    b2v[i] = b2[(wave * 4 + i) * 16 + lm];
#pragma unroll
    for (int r = 0; r < 8; ++r) mem2v[i][r] = 0.f;
  }

  __syncthreads();

#pragma unroll 1
  for (int t = 0; t < T_; ++t) {
    // ---- layer 1: mem1 = beta*mem1 + cur1 - rst1*thr ; spk1 = mem1 > thr ----
    const _Float16* c1p = c1s + er * LDW_ + ec;
    _Float16* sp = spk + er * LDW_ + ec;
#pragma unroll
    for (int j8 = 0; j8 < 16; ++j8) {
      v8h c = *(const v8h*)(c1p + j8 * 8);
      v8h s;
#pragma unroll
      for (int e = 0; e < 8; ++e) {
        float m = mem1v[j8 * 8 + e];
        const float rst = (m > THR_) ? THR_ : 0.f;
        m = BETA_ * m + (float)c[e] - rst;
        mem1v[j8 * 8 + e] = m;
        s[e] = (m > THR_) ? (_Float16)1.f : (_Float16)0.f;
      }
      *(v8h*)(sp + j8 * 8) = s;
    }
    __syncthreads();   // spikes visible to all waves

    // ---- layer 2 GEMM: cur2 = spk1 @ W2^T (A from LDS, B from L2-resident W2) ----
    v8f acc[4];
#pragma unroll
    for (int i = 0; i < 4; ++i) acc[i] = vzero8();

    const _Float16* abase = spk + lm * LDW_ + hi * 8;
#pragma unroll 2
    for (int k0 = 0; k0 < DH_; k0 += 32) {
      ABfrag a;
      a.h[0] = *(const v8h*)(abase + k0);        // K = k0 + hi*8 .. +7
      a.h[1] = *(const v8h*)(abase + k0 + 16);   // K = k0+16 + hi*8 .. +7
#pragma unroll
      for (int i = 0; i < 4; ++i) {
        const int n = (wave * 4 + i) * 16 + lm;
        const _Float16* q = w2h + (size_t)n * DH_ + k0 + hi * 8;
        ABfrag b;
        b.h[0] = *(const v8h*)(q);
        b.h[1] = *(const v8h*)(q + 16);
        acc[i] = __builtin_amdgcn_wmma_f32_16x16x32_f16(false, a.v, false, b.v,
                                                        (short)0, acc[i], false, false);
      }
    }

    // ---- layer 2 membrane update + record (NT stores: keep L2 for W2/cur1) ----
#pragma unroll
    for (int i = 0; i < 4; ++i) {
      const int n = (wave * 4 + i) * 16 + lm;
#pragma unroll
      for (int r = 0; r < 8; ++r) {
        const float c2 = acc[i][r] + b2v[i];
        float m2 = mem2v[i][r];
        const float rst = (m2 > THR_) ? THR_ : 0.f;
        m2 = BETA_ * m2 + c2 - rst;
        mem2v[i][r] = m2;
        const float s2 = (m2 > THR_) ? 1.f : 0.f;
        const size_t o = ((size_t)t * B_ + (size_t)(m0 + r + 8 * hi)) * DOUT_ + n;
        __builtin_nontemporal_store(s2, out_spk + o);
        __builtin_nontemporal_store(m2, out_mem + o);
      }
    }
    __syncthreads();   // all GEMM reads of spk done before next step overwrites
  }
}

extern "C" void kernel_launch(void* const* d_in, const int* in_sizes, int n_in,
                              void* d_out, int out_size, void* d_ws, size_t ws_size,
                              hipStream_t stream) {
  (void)in_sizes; (void)n_in; (void)out_size; (void)ws_size;

  const float* x  = (const float*)d_in[0];
  const float* W1 = (const float*)d_in[1];
  const float* b1 = (const float*)d_in[2];
  const float* W2 = (const float*)d_in[3];
  const float* b2 = (const float*)d_in[4];

  _Float16* xh  = (_Float16*)d_ws;                    // 4096*512
  _Float16* w1h = xh  + (size_t)B_ * DIN_;            // 2048*512
  _Float16* w2h = w1h + (size_t)DH_ * DIN_;           // 512*2048
  _Float16* c1h = w2h + (size_t)DOUT_ * DH_;          // 4096*2048  (24 MB total)

  float* out_spk = (float*)d_out;
  float* out_mem = out_spk + (size_t)T_ * B_ * DOUT_;

  cvt_f16<<<(B_ * DIN_) / 256, 256, 0, stream>>>(x, xh, B_ * DIN_);
  cvt_f16<<<(DH_ * DIN_) / 256, 256, 0, stream>>>(W1, w1h, DH_ * DIN_);
  cvt_f16<<<(DOUT_ * DH_) / 256, 256, 0, stream>>>(W2, w2h, DOUT_ * DH_);
  fc1_kernel<<<B_ / 16, 256, 0, stream>>>(xh, w1h, b1, c1h);
  snn_kernel<<<B_ / 16, 256, 0, stream>>>(c1h, w2h, b2, out_spk, out_mem);
}
